// IFSGenerator_82583631167926
// MI455X (gfx1250) — matrix operand order repaired and checked by reading, not compiled, not run on hardware
//
#include <hip/hip_runtime.h>
#include <hip/hip_bf16.h>
#include <math.h>

typedef __attribute__((ext_vector_type(16))) __bf16 v16bf;
typedef __attribute__((ext_vector_type(8)))  __bf16 v8bf;
typedef __attribute__((ext_vector_type(4)))  __bf16 v4bf;
typedef __attribute__((ext_vector_type(8)))  float  v8f;

#define LATENT 256
#define MID    512
#define TT     5
#define OUTD   35
#define OUTPAD 48
#define ROWS   16
#define WAVES  4
#define LDS_PER_WAVE 73728   // 32K f32 stage + 16K h + 16K t + 8K z

// ---------------------------------------------------------------------------
// Weight swizzle: fp32 row-major (K x N) -> bf16 WMMA B-fragment layout.
// Fragment f = (n_tile*KT + k_tile): 32 lanes x 16 contiguous bf16.
// Lanes 0-15 hold K=k0..k0+15 of column N=lane; lanes 16-31 hold K=k0+16..k0+31.
// ---------------------------------------------------------------------------
__global__ void prep_weights(const float* __restrict__ src, __bf16* __restrict__ dst,
                             int KT, int Ksrc, int Nsrc, int total) {
    int tid = blockIdx.x * blockDim.x + threadIdx.x;
    if (tid >= total) return;
    int e    = tid & 15;
    int lane = (tid >> 4) & 31;
    int f    = tid >> 9;
    int kt   = f % KT;
    int nt   = f / KT;
    int hf   = lane >> 4;
    int m    = lane & 15;
    int k    = kt * 32 + hf * 16 + e;
    int n    = nt * 16 + m;
    float v  = (k < Ksrc && n < Nsrc) ? src[(size_t)k * Nsrc + n] : 0.0f;
    dst[tid] = (__bf16)v;
}

// ---------------------------------------------------------------------------
// A-fragment load from row-major bf16 activations in LDS.
// 16-bit A 16x32: lanes 0-15: K0..15 (row = lane&15);
//                 lanes 16-31: K16..31 (matches B fragment layout).
// => two adjacent 16-byte chunks per lane.
// ---------------------------------------------------------------------------
union AFrag { v16bf v; struct { v8bf lo, hi; } s; };

__device__ __forceinline__ v16bf load_a_frag(const __bf16* act, int stride,
                                             int m, int hf, int kt) {
    const __bf16* p = act + m * stride + kt * 32 + hf * 16;
    AFrag u;
    u.s.lo = *(const v8bf*)(p);
    u.s.hi = *(const v8bf*)(p + 8);
    return u.v;
}

struct BQuad { v16bf b0, b1, b2, b3; };

template <int KT>
__device__ __forceinline__ BQuad load_b_quad(const __bf16* __restrict__ wsw,
                                             int nt0, int kt, int lane) {
    const __bf16* bp = wsw + (nt0 * KT + kt) * 512 + lane * 16;
    BQuad q;
    q.b0 = *(const v16bf*)(bp);
    q.b1 = *(const v16bf*)(bp + KT * 512);
    q.b2 = *(const v16bf*)(bp + 2 * KT * 512);
    q.b3 = *(const v16bf*)(bp + 3 * KT * 512);
    return q;
}

// ---------------------------------------------------------------------------
// 16-row x 512-col GEMM, bf16 WMMA / fp32 accumulate, software-pipelined:
// operands for step k+1 are issued before the 4 WMMAs of step k consume
// theirs, so L2 latency overlaps the matrix pipe instead of stalling it.
// ---------------------------------------------------------------------------
template <int KT>
__device__ __forceinline__ void gemm512(const __bf16* __restrict__ act, int stride,
                                        const __bf16* __restrict__ wsw,
                                        const float* __restrict__ bvec,
                                        float* __restrict__ stageF, int lane) {
    const int hf = lane >> 4, m = lane & 15;
    const v8f vz = {0.f, 0.f, 0.f, 0.f, 0.f, 0.f, 0.f, 0.f};
#pragma unroll 1
    for (int nc = 0; nc < 8; ++nc) {           // 4 N-tiles (64 cols) per chunk
        // warm next chunk's fragment lines while this chunk computes
        if (nc + 1 < 8) {
            const __bf16* np = wsw + ((nc + 1) * 4) * KT * 512 + lane * 16;
            __builtin_prefetch(np, 0, 2);
            __builtin_prefetch(np + KT * 512, 0, 2);
            __builtin_prefetch(np + 2 * KT * 512, 0, 2);
            __builtin_prefetch(np + 3 * KT * 512, 0, 2);
        }
        v8f acc0 = vz, acc1 = vz, acc2 = vz, acc3 = vz;
        v16bf a = load_a_frag(act, stride, m, hf, 0);
        BQuad q = load_b_quad<KT>(wsw, nc * 4, 0, lane);
#pragma unroll
        for (int kt = 0; kt < KT; ++kt) {
            v16bf an;
            BQuad qn;
            if (kt + 1 < KT) {                 // issue next step's loads first
                an = load_a_frag(act, stride, m, hf, kt + 1);
                qn = load_b_quad<KT>(wsw, nc * 4, kt + 1, lane);
            }
            acc0 = __builtin_amdgcn_wmma_f32_16x16x32_bf16(false, a, false, q.b0, (short)0, acc0, false, false);
            acc1 = __builtin_amdgcn_wmma_f32_16x16x32_bf16(false, a, false, q.b1, (short)0, acc1, false, false);
            acc2 = __builtin_amdgcn_wmma_f32_16x16x32_bf16(false, a, false, q.b2, (short)0, acc2, false, false);
            acc3 = __builtin_amdgcn_wmma_f32_16x16x32_bf16(false, a, false, q.b3, (short)0, acc3, false, false);
            if (kt + 1 < KT) { a = an; q = qn; }
        }
        // C/D layout: lane's 8 VGPRs all share N = tile_base + (lane&15);
        // row = r + 8*(lane>>4).
#pragma unroll
        for (int j = 0; j < 4; ++j) {
            v8f acc = (j == 0) ? acc0 : (j == 1) ? acc1 : (j == 2) ? acc2 : acc3;
            int n = (nc * 4 + j) * 16 + m;
            float bias = bvec[n];
#pragma unroll
            for (int r = 0; r < 8; ++r)
                stageF[(r + 8 * hf) * MID + n] = acc[r] + bias;
        }
    }
}

// ---------------------------------------------------------------------------
// LayerNorm over 512 features + activation (+ optional bf16 residual),
// fp32 LDS stage -> bf16 LDS row-major. 2 lanes per row.
// mode: 0 = none, 1 = relu, 2 = silu
// ---------------------------------------------------------------------------
__device__ __forceinline__ void ln_pass(const float* __restrict__ stageF,
                                        __bf16* __restrict__ dst,
                                        const float* __restrict__ g,
                                        const float* __restrict__ be,
                                        int lane, int mode,
                                        const __bf16* __restrict__ resid) {
    const int row = lane >> 1, seg = lane & 1;
    const float* sp = stageF + row * MID + seg * 256;
    float s = 0.f, ss = 0.f;
    for (int i = 0; i < 256; ++i) { float v = sp[i]; s += v; ss += v * v; }
    s  += __shfl_xor(s, 1, 32);
    ss += __shfl_xor(ss, 1, 32);
    const float mu   = s * (1.f / 512.f);
    const float rstd = rsqrtf(ss * (1.f / 512.f) - mu * mu + 1e-5f);
    const int jb = seg * 256;
    for (int i = 0; i < 256; ++i) {
        float v = (sp[i] - mu) * rstd * g[jb + i] + be[jb + i];
        if (mode == 1)      v = fmaxf(v, 0.f);
        else if (mode == 2) v = v / (1.f + __expf(-v));
        if (resid) v += (float)resid[row * MID + jb + i];
        dst[row * MID + jb + i] = (__bf16)v;
    }
}

// ---------------------------------------------------------------------------
// Fused MLP + LN + softmax + closed-form clamped 2x2 SVD.
// 4 independent waves per block (one per SIMD32 of the WGP) so all four
// stream the same weight fragments in phase -> WGP$/L1 reuse. Each wave owns
// a private 72KB LDS region; no barriers anywhere.
// ---------------------------------------------------------------------------
__global__ __launch_bounds__(WAVES * 32) void ifs_main(
    const float*  __restrict__ z,
    const __bf16* __restrict__ w1sw, const float* __restrict__ b1,
    const float*  __restrict__ g1,   const float* __restrict__ be1,
    const __bf16* __restrict__ wasw, const float* __restrict__ ba,
    const float*  __restrict__ ga,   const float* __restrict__ bea,
    const __bf16* __restrict__ wbsw, const float* __restrict__ bb,
    const float*  __restrict__ gb,   const float* __restrict__ beb,
    const __bf16* __restrict__ wosw, const float* __restrict__ bout,
    float* __restrict__ out, long long Btot) {
    extern __shared__ __align__(16) char smem[];
    const int tid  = threadIdx.x;
    const int wave = tid >> 5;
    const int lane = tid & 31;
    char* wbase = smem + (size_t)wave * LDS_PER_WAVE;
    float*  stageF = (float*)wbase;                       // 32768 B
    __bf16* hbuf   = (__bf16*)(wbase + 32768);            // 16384 B
    __bf16* tbuf   = (__bf16*)(wbase + 49152);            // 16384 B
    __bf16* zbuf   = (__bf16*)(wbase + 65536);            //  8192 B

    const int hf = lane >> 4, m = lane & 15;
    const size_t rowBase = ((size_t)blockIdx.x * WAVES + wave) * ROWS;

    // ---- stage z (fp32 global) -> bf16 LDS, row-major [16][256] ----
    {
        const float4* zg = (const float4*)(z + rowBase * LATENT);
#pragma unroll 4
        for (int i = 0; i < 32; ++i) {                    // 1024 float4 / 32 lanes
            int idx = i * 32 + lane;
            float4 v = zg[idx];
            v4bf o = {(__bf16)v.x, (__bf16)v.y, (__bf16)v.z, (__bf16)v.w};
            *(v4bf*)(zbuf + idx * 4) = o;
        }
    }

    // ---- layer 1: h = relu(LN(z @ W1 + b1)) ----
    gemm512<8>(zbuf, LATENT, w1sw, b1, stageF, lane);
    ln_pass(stageF, hbuf, g1, be1, lane, /*relu*/1, nullptr);

    // ---- layer a: t = silu(LN(h @ Wa + ba)) ----
    gemm512<16>(hbuf, MID, wasw, ba, stageF, lane);
    ln_pass(stageF, tbuf, ga, bea, lane, /*silu*/2, nullptr);

    // ---- layer b: h2 = h + LN(t @ Wb + bb)  (h2 overwrites tbuf) ----
    gemm512<16>(tbuf, MID, wbsw, bb, stageF, lane);
    ln_pass(stageF, tbuf, gb, beb, lane, /*none*/0, hbuf);

    // ---- output layer: x = h2 @ Wout + bout (35 cols padded to 48) ----
    {
        const v8f vz = {0.f, 0.f, 0.f, 0.f, 0.f, 0.f, 0.f, 0.f};
        v8f acc0 = vz, acc1 = vz, acc2 = vz;
        v16bf a = load_a_frag(tbuf, MID, m, hf, 0);
        const __bf16* bp0 = wosw + lane * 16;
        v16bf b0 = *(const v16bf*)(bp0);
        v16bf b1 = *(const v16bf*)(bp0 + 16 * 512);
        v16bf b2 = *(const v16bf*)(bp0 + 32 * 512);
#pragma unroll
        for (int kt = 0; kt < 16; ++kt) {
            v16bf an, bn0, bn1, bn2;
            if (kt + 1 < 16) {
                an = load_a_frag(tbuf, MID, m, hf, kt + 1);
                const __bf16* bp = wosw + (kt + 1) * 512 + lane * 16;
                bn0 = *(const v16bf*)(bp);
                bn1 = *(const v16bf*)(bp + 16 * 512);
                bn2 = *(const v16bf*)(bp + 32 * 512);
            }
            acc0 = __builtin_amdgcn_wmma_f32_16x16x32_bf16(false, a, false, b0, (short)0, acc0, false, false);
            acc1 = __builtin_amdgcn_wmma_f32_16x16x32_bf16(false, a, false, b1, (short)0, acc1, false, false);
            acc2 = __builtin_amdgcn_wmma_f32_16x16x32_bf16(false, a, false, b2, (short)0, acc2, false, false);
            if (kt + 1 < 16) { a = an; b0 = bn0; b1 = bn1; b2 = bn2; }
        }
#pragma unroll
        for (int j = 0; j < 3; ++j) {
            v8f acc = (j == 0) ? acc0 : (j == 1) ? acc1 : acc2;
            int n = j * 16 + m;
            float bias = (n < OUTD) ? bout[n] : 0.f;
#pragma unroll
            for (int r = 0; r < 8; ++r)
                stageF[(r + 8 * hf) * OUTPAD + n] = acc[r] + bias;
        }
    }

    // ---- per-row epilogue: softmax(5) + clamped 2x2 SVD x5 ----
    if (lane < ROWS) {
        const float* xr = stageF + lane * OUTPAD;
        const size_t rg = rowBase + lane;

        // softmax over logits x[30..34]
        float lg[TT];
        float mx = -1e30f;
#pragma unroll
        for (int i = 0; i < TT; ++i) { lg[i] = xr[30 + i]; mx = fmaxf(mx, lg[i]); }
        float sum = 0.f;
#pragma unroll
        for (int i = 0; i < TT; ++i) { lg[i] = __expf(lg[i] - mx); sum += lg[i]; }
        float inv = 1.f / sum;
        float* op = out + (size_t)Btot * 30 + rg * TT;
#pragma unroll
        for (int i = 0; i < TT; ++i) op[i] = lg[i] * inv;

        float* oa = out + rg * 30;
#pragma unroll
        for (int t = 0; t < TT; ++t) {
            float a  = xr[t * 6 + 0], b  = xr[t * 6 + 1], tx = xr[t * 6 + 2];
            float c  = xr[t * 6 + 3], d  = xr[t * 6 + 4], ty = xr[t * 6 + 5];
            // closed-form SVD of [[a,b],[c,d]]
            float E = 0.5f * (a + d), F = 0.5f * (a - d);
            float G = 0.5f * (c + b), H = 0.5f * (c - b);
            float Q = sqrtf(E * E + H * H), R = sqrtf(F * F + G * G);
            float s1 = Q + R, s2 = Q - R;                 // |s2| = 2nd singular value
            float a1 = atan2f(G, F), a2 = atan2f(H, E);
            float th = 0.5f * (a2 - a1), ph = 0.5f * (a2 + a1);
            float sth, cth, sph, cph;
            __sincosf(th, &sth, &cth);
            __sincosf(ph, &sph, &cph);
            float s1c = fminf(s1, 0.9f);
            float s2c = copysignf(fminf(fabsf(s2), 0.9f), s2);
            // M2 = U * diag(s1c,s2c) * Vt, U=[[cph,-sph],[sph,cph]], Vt=[[cth,sth],[-sth,cth]]
            float u00 = cph * s1c, u01 = -sph * s2c;
            float u10 = sph * s1c, u11 =  cph * s2c;
            oa[t * 6 + 0] = u00 * cth - u01 * sth;
            oa[t * 6 + 1] = u00 * sth + u01 * cth;
            oa[t * 6 + 2] = tx;
            oa[t * 6 + 3] = u10 * cth - u11 * sth;
            oa[t * 6 + 4] = u10 * sth + u11 * cth;
            oa[t * 6 + 5] = ty;
        }
    }
}

// ---------------------------------------------------------------------------
extern "C" void kernel_launch(void* const* d_in, const int* in_sizes, int n_in,
                              void* d_out, int out_size, void* d_ws, size_t ws_size,
                              hipStream_t stream) {
    const float* z    = (const float*)d_in[0];
    const float* W1   = (const float*)d_in[1];
    const float* b1   = (const float*)d_in[2];
    const float* g1   = (const float*)d_in[3];
    const float* be1  = (const float*)d_in[4];
    const float* Wa   = (const float*)d_in[5];
    const float* ba   = (const float*)d_in[6];
    const float* ga   = (const float*)d_in[7];
    const float* bea  = (const float*)d_in[8];
    const float* Wb   = (const float*)d_in[9];
    const float* bb   = (const float*)d_in[10];
    const float* gb   = (const float*)d_in[11];
    const float* beb  = (const float*)d_in[12];
    const float* Wout = (const float*)d_in[13];
    const float* bout = (const float*)d_in[14];

    long long Btot = (long long)in_sizes[0] / LATENT;

    // workspace: swizzled bf16 weights
    __bf16* w1sw = (__bf16*)d_ws;            // 32 NT * 8 KT * 512  = 131072
    __bf16* wasw = w1sw + 32 * 8 * 512;      // 32 NT * 16 KT * 512 = 262144
    __bf16* wbsw = wasw + 32 * 16 * 512;     // 262144
    __bf16* wosw = wbsw + 32 * 16 * 512;     //  3 NT * 16 KT * 512 = 24576

    {
        int t1 = 32 * 8 * 512;
        prep_weights<<<(t1 + 255) / 256, 256, 0, stream>>>(W1, w1sw, 8, LATENT, MID, t1);
        int t2 = 32 * 16 * 512;
        prep_weights<<<(t2 + 255) / 256, 256, 0, stream>>>(Wa, wasw, 16, MID, MID, t2);
        prep_weights<<<(t2 + 255) / 256, 256, 0, stream>>>(Wb, wbsw, 16, MID, MID, t2);
        int t3 = 3 * 16 * 512;
        prep_weights<<<(t3 + 255) / 256, 256, 0, stream>>>(Wout, wosw, 16, MID, OUTD, t3);
    }

    size_t shmem = (size_t)WAVES * LDS_PER_WAVE;   // 288 KB per workgroup (4 waves)
    int grid = (int)(Btot / (ROWS * WAVES));
    ifs_main<<<grid, WAVES * 32, shmem, stream>>>(z, w1sw, b1, g1, be1,
                                                  wasw, ba, ga, bea,
                                                  wbsw, bb, gb, beb,
                                                  wosw, bout, (float*)d_out, Btot);
}
